// Head_15453292331576
// MI455X (gfx1250) — matrix operand order repaired and checked by reading, not compiled, not run on hardware
//
#include <hip/hip_runtime.h>
#include <hip/hip_bf16.h>

// Attention head: B=4, T=4096, C=1024, H=64.
// Pass 0: fp32 -> bf16 streaming conversion of x and W (HBM-bound; the whole
//         problem's roofline is the one 64MB pass over x at 23.3 TB/s).
// Pass 1: QKV projection, pure bf16 WMMA, all fragment loads b128.
// Pass 2: flash attention (online softmax), WMMA for S=QK^T and O+=PV.

#define B_  4
#define T_  4096
#define C_  1024
#define H_  64
#define BT_ (B_ * T_)

typedef __attribute__((ext_vector_type(16))) __bf16 bf16x16;
typedef __attribute__((ext_vector_type(2)))  __bf16 bf16x2;
typedef __attribute__((ext_vector_type(8)))  float  f32x8;

// Fragment viewed as two 16B chunks (b128 loads) or 8 packed dwords.
union FragB { bf16x16 v; uint4 q[2]; unsigned u[8]; };

__device__ inline unsigned pack2bf(float lo, float hi) {
#if __has_builtin(__builtin_amdgcn_cvt_pk_bf16_f32)
  union { bf16x2 v; unsigned u; } t;
  t.v = __builtin_amdgcn_cvt_pk_bf16_f32(lo, hi);   // 1 VALU op / 2 values
  return t.u;
#else
  union { float f; unsigned u; } a, b; a.f = lo; b.f = hi;
  unsigned rl = (a.u + 0x7FFFu + ((a.u >> 16) & 1u)) >> 16;   // RNE
  unsigned rh = (b.u + 0x7FFFu + ((b.u >> 16) & 1u)) >> 16;
  return (rl & 0xFFFFu) | (rh << 16);
#endif
}
__device__ inline unsigned short f2bfbits(float f) {
  return (unsigned short)(pack2bf(f, 0.0f) & 0xFFFFu);
}
__device__ inline float rmax16(float v) {
  v = fmaxf(v, __shfl_xor(v, 1, 32));
  v = fmaxf(v, __shfl_xor(v, 2, 32));
  v = fmaxf(v, __shfl_xor(v, 4, 32));
  v = fmaxf(v, __shfl_xor(v, 8, 32));
  return v;
}
__device__ inline float rsum16(float v) {
  v += __shfl_xor(v, 1, 32);
  v += __shfl_xor(v, 2, 32);
  v += __shfl_xor(v, 4, 32);
  v += __shfl_xor(v, 8, 32);
  return v;
}

// ---------------------------------------------------------------------------
// Kernel 0: streaming fp32 -> bf16 (8 floats / thread, b128 in + b128 out).
// ---------------------------------------------------------------------------
__global__ __launch_bounds__(256) void cvt_bf16_kernel(
    const float* __restrict__ src, unsigned short* __restrict__ dst, int n8)
{
  int i = blockIdx.x * 256 + threadIdx.x;
  if (i >= n8) return;
  size_t base = (size_t)i * 8;
  float4 f0 = *reinterpret_cast<const float4*>(src + base);
  float4 f1 = *reinterpret_cast<const float4*>(src + base + 4);
  uint4 o;
  o.x = pack2bf(f0.x, f0.y);
  o.y = pack2bf(f0.z, f0.w);
  o.z = pack2bf(f1.x, f1.y);
  o.w = pack2bf(f1.z, f1.w);
  *reinterpret_cast<uint4*>(dst + base) = o;
}

// ---------------------------------------------------------------------------
// Kernel 1: q/k/v = x @ W^T  (M=B*T, N=64 each, K=1024), all-bf16 WMMA.
// One wave per 16-row tile; A fragment reused across 12 accumulators and
// double-buffered across K-steps. Every fragment = two b128 loads.
// ---------------------------------------------------------------------------
__global__ __launch_bounds__(256) void qkv_proj_kernel(
    const unsigned short* __restrict__ xbf,   // [B*T][C] bf16
    const unsigned short* __restrict__ wbf,   // [3][64][C] bf16 (q,k,v)
    unsigned short* __restrict__ qkv)         // [3][B*T][64] bf16
{
  const int lane = threadIdx.x & 31;
  const int wave = threadIdx.x >> 5;
  const int rowtile = blockIdx.x * 8 + wave;
  const int row0 = rowtile * 16;
  const int n    = lane & 15;
  const int half = lane >> 4;

  f32x8 acc[12];
#pragma unroll
  for (int t = 0; t < 12; ++t)
#pragma unroll
    for (int i = 0; i < 8; ++i) acc[t][i] = 0.0f;

  const unsigned short* xr = xbf + (size_t)(row0 + n) * C_;  // A row = lane&15

  // A 16x32 layout: K[half*16 .. +15] contiguous
  FragB a;
  a.q[0] = *reinterpret_cast<const uint4*>(xr + half * 16);
  a.q[1] = *reinterpret_cast<const uint4*>(xr + half * 16 + 8);

  for (int k0 = 0; k0 < C_; k0 += 32) {
    FragB anext;
    if (k0 + 32 < C_) {  // issue next A loads before this step's WMMAs
      anext.q[0] = *reinterpret_cast<const uint4*>(xr + k0 + 32 + half * 16);
      anext.q[1] = *reinterpret_cast<const uint4*>(xr + k0 + 40 + half * 16);
    }
#pragma unroll
    for (int w = 0; w < 3; ++w) {
#pragma unroll
      for (int nt = 0; nt < 4; ++nt) {
        const unsigned short* wr = wbf + (size_t)(w * 64 + nt * 16 + n) * C_;
        FragB bm;  // B 32x16: K[half*16 .. +15] contiguous
        bm.q[0] = *reinterpret_cast<const uint4*>(wr + k0 + half * 16);
        bm.q[1] = *reinterpret_cast<const uint4*>(wr + k0 + half * 16 + 8);
        acc[w * 4 + nt] = __builtin_amdgcn_wmma_f32_16x16x32_bf16(
            false, a.v, false, bm.v, (short)0, acc[w * 4 + nt], false, false);
      }
    }
    a = anext;
  }
  // C/D layout: VGPR r -> M = r + 8*half, N = lane&15
#pragma unroll
  for (int w = 0; w < 3; ++w)
#pragma unroll
    for (int nt = 0; nt < 4; ++nt)
#pragma unroll
      for (int r = 0; r < 8; ++r) {
        int row = row0 + r + 8 * half;
        qkv[(size_t)w * BT_ * H_ + (size_t)row * H_ + nt * 16 + n] =
            f2bfbits(acc[w * 4 + nt][r]);
      }
}

// ---------------------------------------------------------------------------
// Kernel 2: flash attention. Block = 8 waves x 16 query rows = 128 rows.
// K-tile [32 keys][64] and V-tile transposed [64][32 keys] staged in LDS.
// ---------------------------------------------------------------------------
__global__ __launch_bounds__(256) void attn_kernel(
    const unsigned short* __restrict__ qbf,
    const unsigned short* __restrict__ kbf,
    const unsigned short* __restrict__ vbf,
    float* __restrict__ out)
{
  __shared__ unsigned short Ks[32 * 64];        // [key][h]
  __shared__ unsigned short Vt[64 * 32];        // [h][key]
  __shared__ unsigned short Ps[8][16 * 32];     // per-wave P staging (C->A relayout)

  const int tid  = threadIdx.x;
  const int lane = tid & 31;
  const int wave = tid >> 5;
  const int n    = lane & 15;
  const int half = lane >> 4;
  const int b    = blockIdx.y;
  const int qbase_blk = blockIdx.x * 128;
  const int qrow0   = qbase_blk + wave * 16;
  const int wq_max  = qrow0 + 15;
  const int blk_qmax = qbase_blk + 127;
  const float scale = 0.03125f;                 // C^-0.5 = 1/32

  // Q A-fragments: two 16x32 frags over H=64, two b128 loads each.
  const unsigned short* qb = qbf + ((size_t)b * T_ + qrow0) * H_ + (size_t)n * H_;
  FragB qa[2];
#pragma unroll
  for (int f = 0; f < 2; ++f) {
    qa[f].q[0] = *reinterpret_cast<const uint4*>(qb + f * 32 + half * 16);
    qa[f].q[1] = *reinterpret_cast<const uint4*>(qb + f * 32 + half * 16 + 8);
  }

  float m_r[8], l_r[8];
  f32x8 o[4];
#pragma unroll
  for (int r = 0; r < 8; ++r) { m_r[r] = -1e30f; l_r[r] = 0.0f; }
#pragma unroll
  for (int f = 0; f < 4; ++f)
#pragma unroll
    for (int i = 0; i < 8; ++i) o[f][i] = 0.0f;

  // Cooperative staging geometry: 256 threads = 32 keys x 8 chunks of 8 h.
  const int skey = tid >> 3;
  const int sc8  = (tid & 7) * 8;

  const int nkb = (blk_qmax >> 5) + 1;
  for (int kb = 0; kb < nkb; ++kb) {
    const int kbase = kb * 32;
    {
      size_t g = ((size_t)b * T_ + kbase + skey) * H_ + sc8;
      // K tile row-major: straight b128 copy.
      *reinterpret_cast<uint4*>(&Ks[skey * 64 + sc8]) =
          *reinterpret_cast<const uint4*>(&kbf[g]);
      // V tile transposed: b128 load, 8 b16 scatters.
      union { uint4 q; unsigned short s[8]; } vv;
      vv.q = *reinterpret_cast<const uint4*>(&vbf[g]);
#pragma unroll
      for (int e = 0; e < 8; ++e) Vt[(sc8 + e) * 32 + skey] = vv.s[e];
      // Prefetch next key block while this one is consumed.
      if (kb + 1 < nkb) {
        __builtin_prefetch(&kbf[g + (size_t)32 * H_], 0, 0);
        __builtin_prefetch(&vbf[g + (size_t)32 * H_], 0, 0);
      }
    }
    __syncthreads();

    if (kbase <= wq_max) {
      // ---- S = Q K^T (two 16x16 tiles over 32 keys; K-dim = H = 64) ----
      f32x8 s0 = {0, 0, 0, 0, 0, 0, 0, 0};
      f32x8 s1 = {0, 0, 0, 0, 0, 0, 0, 0};
#pragma unroll
      for (int t = 0; t < 2; ++t) {            // key sub-tile
#pragma unroll
        for (int seg = 0; seg < 2; ++seg) {    // h segment of 32
          FragB bk;                            // B 32x16: K-dim=h, N=key
          const unsigned short* kr = &Ks[(t * 16 + n) * 64 + seg * 32 + half * 16];
          bk.q[0] = *reinterpret_cast<const uint4*>(kr);
          bk.q[1] = *reinterpret_cast<const uint4*>(kr + 8);
          if (t == 0)
            s0 = __builtin_amdgcn_wmma_f32_16x16x32_bf16(
                false, qa[seg].v, false, bk.v, (short)0, s0, false, false);
          else
            s1 = __builtin_amdgcn_wmma_f32_16x16x32_bf16(
                false, qa[seg].v, false, bk.v, (short)0, s1, false, false);
        }
      }

      // ---- scale + causal mask + online softmax ----
      float alpha[8];
#pragma unroll
      for (int r = 0; r < 8; ++r) {
        int qi  = qrow0 + r + 8 * half;
        float v0 = s0[r] * scale;
        float v1 = s1[r] * scale;
        if (kbase + n      > qi) v0 = -1e30f;
        if (kbase + 16 + n > qi) v1 = -1e30f;
        float mx = rmax16(fmaxf(v0, v1));
        float mn = fmaxf(m_r[r], mx);
        float al = __expf(m_r[r] - mn);
        float p0 = __expf(v0 - mn);
        float p1 = __expf(v1 - mn);
        s0[r] = p0; s1[r] = p1;
        l_r[r] = l_r[r] * al + rsum16(p0 + p1);
        m_r[r] = mn;
        alpha[r] = al;
      }
#pragma unroll
      for (int f = 0; f < 4; ++f)
#pragma unroll
        for (int r = 0; r < 8; ++r) o[f][r] *= alpha[r];

      // ---- relayout P: C-layout -> A-layout via per-wave LDS scratch ----
      unsigned short* Pw = Ps[wave];
#pragma unroll
      for (int r = 0; r < 8; ++r) {
        int m = r + 8 * half;
        Pw[m * 32 + n]      = f2bfbits(s0[r]);
        Pw[m * 32 + 16 + n] = f2bfbits(s1[r]);
      }
      asm volatile("s_wait_dscnt 0" ::: "memory");  // LDS RAW within wave
      FragB pa;                                     // A 16x32: M=q row, K=key
      pa.q[0] = *reinterpret_cast<const uint4*>(&Pw[n * 32 + half * 16]);
      pa.q[1] = *reinterpret_cast<const uint4*>(&Pw[n * 32 + half * 16 + 8]);

      // ---- O += P V (4 h-tiles of 16) ----
#pragma unroll
      for (int f = 0; f < 4; ++f) {
        FragB bv;                                   // B 32x16: K-dim=key, N=h
        const unsigned short* vr = &Vt[(f * 16 + n) * 32 + half * 16];
        bv.q[0] = *reinterpret_cast<const uint4*>(vr);
        bv.q[1] = *reinterpret_cast<const uint4*>(vr + 8);
        o[f] = __builtin_amdgcn_wmma_f32_16x16x32_bf16(
            false, pa.v, false, bv.v, (short)0, o[f], false, false);
      }
    }
    __syncthreads();
  }

  // ---- write O / l (fp32 output) ----
#pragma unroll
  for (int f = 0; f < 4; ++f)
#pragma unroll
    for (int r = 0; r < 8; ++r) {
      int row = qrow0 + r + 8 * half;
      out[((size_t)b * T_ + row) * H_ + f * 16 + n] = o[f][r] / l_r[r];
    }
}

extern "C" void kernel_launch(void* const* d_in, const int* in_sizes, int n_in,
                              void* d_out, int out_size, void* d_ws, size_t ws_size,
                              hipStream_t stream) {
  const float* x  = (const float*)d_in[0];
  const float* Wk = (const float*)d_in[1];
  const float* Wq = (const float*)d_in[2];
  const float* Wv = (const float*)d_in[3];
  float* out = (float*)d_out;

  // Workspace layout (bf16 = unsigned short):
  //   xbf [B*T][C]          : 32 MB
  //   wbf [3][64][C] (q,k,v): 384 KB
  //   qkv [3][B*T][64]      : 6 MB
  unsigned short* xbf = (unsigned short*)d_ws;
  unsigned short* wbf = xbf + (size_t)BT_ * C_;
  unsigned short* qkv = wbf + (size_t)3 * H_ * C_;
  unsigned short* qbf = qkv;
  unsigned short* kbf = qkv + (size_t)BT_ * H_;
  unsigned short* vbf = kbf + (size_t)BT_ * H_;

  // Pass 0: convert x and weights to bf16 (streaming, HBM-bound).
  {
    int n8 = (BT_ * C_) / 8;  // 2,097,152
    cvt_bf16_kernel<<<dim3(n8 / 256), 256, 0, stream>>>(x, xbf, n8);
    int w8 = (H_ * C_) / 8;   // 8,192
    cvt_bf16_kernel<<<dim3(w8 / 256), 256, 0, stream>>>(Wq, wbf + 0 * H_ * C_, w8);
    cvt_bf16_kernel<<<dim3(w8 / 256), 256, 0, stream>>>(Wk, wbf + 1 * H_ * C_, w8);
    cvt_bf16_kernel<<<dim3(w8 / 256), 256, 0, stream>>>(Wv, wbf + 2 * H_ * C_, w8);
  }

  // Pass 1: projection. (B*T/16) row tiles, 8 waves per block.
  qkv_proj_kernel<<<dim3((BT_ / 16) / 8), 256, 0, stream>>>(xbf, wbf, qkv);

  // Pass 2: attention.
  attn_kernel<<<dim3(T_ / 128, B_), 256, 0, stream>>>(qbf, kbf, vbf, out);
}
